// LocalAttention_2714419331379
// MI455X (gfx1250) — matrix operand order repaired and checked by reading, not compiled, not run on hardware
//
#include <hip/hip_runtime.h>
#include <hip/hip_bf16.h>

typedef __attribute__((ext_vector_type(16))) _Float16 v16h;
typedef __attribute__((ext_vector_type(8)))  float    v8f;

// Local causal attention, bucket=64, window=128 (prev bucket + current), E=64.
// One workgroup per (bh, bucket) tile; 4 waves x 32 lanes; each wave computes a
// 16-row query strip with v_wmma_f32_16x16x32_f16.
__global__ __launch_bounds__(128)
void local_attn_wmma_kernel(const float* __restrict__ q,
                            const float* __restrict__ k,
                            const float* __restrict__ v,
                            float* __restrict__ out,
                            int T) {
  constexpr int E   = 64;   // head dim
  constexpr int BUK = 64;   // bucket size
  constexpr int WIN = 128;  // key window
  constexpr int QS  = 72;   // padded LDS stride (144B, 16B multiple)
  constexpr int PS  = 136;  // padded LDS stride (272B, 16B multiple)

  __shared__ _Float16 sQ [BUK][QS];   // Q tile, row-major (f16)
  __shared__ _Float16 sK [WIN][QS];   // K window, row-major (f16)
  __shared__ _Float16 sVt[E  ][PS];   // V window, TRANSPOSED: [e][j]
  __shared__ _Float16 sP [BUK][PS];   // softmax probs (f16), [m][j]

  const int tid    = threadIdx.x;         // 0..127
  const int bucket = blockIdx.x;          // 0..T/64-1
  const int bh     = blockIdx.y;          // merged batch*heads

  const size_t qbase = ((size_t)bh * (size_t)T + (size_t)bucket * BUK) * E;

  // ---------------- stage Q ----------------
  {
    const float4* qg = reinterpret_cast<const float4*>(q + qbase); // tile is contiguous
    #pragma unroll
    for (int i = tid; i < BUK * E / 4; i += 128) {
      float4 f = qg[i];
      int row = i >> 4, c = (i & 15) << 2;
      sQ[row][c + 0] = (_Float16)f.x;
      sQ[row][c + 1] = (_Float16)f.y;
      sQ[row][c + 2] = (_Float16)f.z;
      sQ[row][c + 3] = (_Float16)f.w;
    }
  }
  // ---------------- stage K / V window ----------------
  {
    #pragma unroll
    for (int i = tid; i < WIN * E / 4; i += 128) {
      int j = i >> 4, c = (i & 15) << 2;      // window row, col
      int grow = bucket * BUK - BUK + j;      // global seq row (may be <0 => pad)
      float4 fk = make_float4(0.f, 0.f, 0.f, 0.f);
      float4 fv = make_float4(0.f, 0.f, 0.f, 0.f);
      if (grow >= 0) {
        size_t o4 = (((size_t)bh * (size_t)T + (size_t)grow) * E + c) >> 2;
        fk = reinterpret_cast<const float4*>(k)[o4];
        fv = reinterpret_cast<const float4*>(v)[o4];
      }
      sK[j][c + 0] = (_Float16)fk.x;
      sK[j][c + 1] = (_Float16)fk.y;
      sK[j][c + 2] = (_Float16)fk.z;
      sK[j][c + 3] = (_Float16)fk.w;
      sVt[c + 0][j] = (_Float16)fv.x;
      sVt[c + 1][j] = (_Float16)fv.y;
      sVt[c + 2][j] = (_Float16)fv.z;
      sVt[c + 3][j] = (_Float16)fv.w;
    }
  }
  __syncthreads();

  const int lane = tid & 31;
  const int wv   = tid >> 5;        // wave 0..3
  const int l16  = lane & 15;
  const int half = lane >> 4;       // 0: lanes 0-15, 1: lanes 16-31
  const int m0   = wv * 16;         // query strip base row

  // ---- A operands: Q strip, 2 k-chunks of 32 (contiguous k order, same as B) ----
  v16h aq[2];
  #pragma unroll
  for (int c = 0; c < 2; ++c) {
    const _Float16* r = &sQ[m0 + l16][0];
    const int e0 = c * 32 + half * 16;      // K 0..7  (half 0) / 16..23 (half 1)
    const int e1 = c * 32 + half * 16 + 8;  // K 8..15 (half 0) / 24..31 (half 1)
    #pragma unroll
    for (int t = 0; t < 8; ++t) { aq[c][t] = r[e0 + t]; aq[c][8 + t] = r[e1 + t]; }
  }

  // ---- S = Q * K^T : 8 tiles of 16x16 over 128-key window ----
  v8f S[8];
  #pragma unroll
  for (int nt = 0; nt < 8; ++nt) {
    v8f acc = {};
    #pragma unroll
    for (int c = 0; c < 2; ++c) {
      v16h bk;  // B layout: lane = key column, 16 contiguous e-values (+16 for upper half)
      const _Float16* kr = &sK[nt * 16 + l16][c * 32 + half * 16];
      #pragma unroll
      for (int t = 0; t < 16; ++t) bk[t] = kr[t];
      acc = __builtin_amdgcn_wmma_f32_16x16x32_f16(false, aq[c], false, bk,
                                                   (short)0, acc, false, false);
    }
    S[nt] = acc;
  }

  // ---- scale + causal/pad mask + row max ----
  const float scale = 0.125f;  // 64^-0.5
  float rmax[8], rsum[8], inv[8];
  #pragma unroll
  for (int r = 0; r < 8; ++r) rmax[r] = -3.0e38f;
  #pragma unroll
  for (int nt = 0; nt < 8; ++nt) {
    #pragma unroll
    for (int r = 0; r < 8; ++r) {
      const int m = m0 + r + 8 * half;      // query row within bucket
      const int n = nt * 16 + l16;          // window key col (0..127)
      float val = S[nt][r] * scale;
      const bool masked = (n - BUK > m) || (bucket == 0 && n < BUK);
      val = masked ? -3.0e38f : val;
      S[nt][r] = val;
      rmax[r] = fmaxf(rmax[r], val);
    }
  }
  #pragma unroll
  for (int r = 0; r < 8; ++r) {
    float mx = rmax[r];
    mx = fmaxf(mx, __shfl_xor(mx, 1, 32));
    mx = fmaxf(mx, __shfl_xor(mx, 2, 32));
    mx = fmaxf(mx, __shfl_xor(mx, 4, 32));
    mx = fmaxf(mx, __shfl_xor(mx, 8, 32));
    rmax[r] = mx;
    rsum[r] = 0.f;
  }

  // ---- P = exp(S - max); write f16 to LDS for layout conversion ----
  #pragma unroll
  for (int nt = 0; nt < 8; ++nt) {
    #pragma unroll
    for (int r = 0; r < 8; ++r) {
      float p = __expf(S[nt][r] - rmax[r]);
      rsum[r] += p;
      sP[m0 + r + 8 * half][nt * 16 + l16] = (_Float16)p;
    }
  }
  #pragma unroll
  for (int r = 0; r < 8; ++r) {
    float s = rsum[r];
    s += __shfl_xor(s, 1, 32);
    s += __shfl_xor(s, 2, 32);
    s += __shfl_xor(s, 4, 32);
    s += __shfl_xor(s, 8, 32);
    inv[r] = 1.0f / s;
  }
  __syncthreads();

  // ---- A operands: P strip, 4 k-chunks of 32 keys ----
  v16h ap[4];
  #pragma unroll
  for (int kc = 0; kc < 4; ++kc) {
    const _Float16* pr = &sP[m0 + l16][0];
    const int j0 = kc * 32 + half * 16;
    const int j1 = kc * 32 + half * 16 + 8;
    #pragma unroll
    for (int t = 0; t < 8; ++t) { ap[kc][t] = pr[j0 + t]; ap[kc][8 + t] = pr[j1 + t]; }
  }

  // ---- O = P * V (4 e-tiles), normalize, store ----
  float* og = out + qbase;
  #pragma unroll
  for (int et = 0; et < 4; ++et) {
    v8f o = {};
    #pragma unroll
    for (int kc = 0; kc < 4; ++kc) {
      v16h bv;  // B layout: lane = e column, 16 contiguous key rows from sVt
      const _Float16* vr = &sVt[et * 16 + l16][kc * 32 + half * 16];
      #pragma unroll
      for (int t = 0; t < 16; ++t) bv[t] = vr[t];
      o = __builtin_amdgcn_wmma_f32_16x16x32_f16(false, ap[kc], false, bv,
                                                 (short)0, o, false, false);
    }
    #pragma unroll
    for (int r = 0; r < 8; ++r) {
      const int m = m0 + r + 8 * half;
      og[(size_t)m * E + et * 16 + l16] = o[r] * inv[r];
    }
  }
}

extern "C" void kernel_launch(void* const* d_in, const int* in_sizes, int n_in,
                              void* d_out, int out_size, void* d_ws, size_t ws_size,
                              hipStream_t stream) {
  const float* q = (const float*)d_in[0];
  const float* k = (const float*)d_in[1];
  const float* v = (const float*)d_in[2];
  float* out = (float*)d_out;

  const int T = 4096, E = 64, BUK = 64;
  const int bh = in_sizes[0] / (T * E);    // 4*16 = 64 merged batch*heads
  dim3 grid(T / BUK, bh);                  // (buckets, bh)
  local_attn_wmma_kernel<<<grid, 128, 0, stream>>>(q, k, v, out, T);
}
